// NeRFLE_30743375905035
// MI455X (gfx1250) — matrix operand order repaired and checked by reading, not compiled, not run on hardware
//
#include <hip/hip_runtime.h>

// ---------------------------------------------------------------------------
// Fused NeRF-style pipeline for MI455X (gfx1250, wave32, WMMA + TDM).
//   - 8192 rays x 64 samples; two 128-wide MLPs (skip concat) + volume render.
//   - One wave == one ray (64 points = 4 WMMA M-tiles of 16).
//   - Activations ping-pong through LDS in f16 (pitch 136); skip-concat handled
//     as a second K-pass over small aux input buffers (IN1/IN2).
//   - Weights pre-transposed to [N][Kpad] f16 in d_ws; per-layer tiles streamed
//     into a DOUBLE-BUFFERED LDS stage by the Tensor Data Mover (overlapped
//     with compute), falling back to manual staging if the builtin is absent.
//   - Buffer selection uses ternaries (NOT pointer arrays) so InferAddressSpaces
//     keeps every activation access in LDS (ds_load/ds_store, no flat ops).
//   - All GEMMs: v_wmma_f32_16x16x32_f16.
// ---------------------------------------------------------------------------

typedef _Float16 h16;
typedef _Float16 v16h __attribute__((ext_vector_type(16)));
typedef _Float16 v8h  __attribute__((ext_vector_type(8)));
typedef float    v8f  __attribute__((ext_vector_type(8)));
typedef unsigned int u32x4 __attribute__((ext_vector_type(4)));
typedef int          i32x4 __attribute__((ext_vector_type(4)));
typedef int          i32x8 __attribute__((ext_vector_type(8)));

#define ACT_PITCH       136                  // halves per row, ping/pong buffers
#define IN1_PITCH       32                   // MLP1 raw input (x, 3 -> pad 32)
#define IN2_PITCH       96                   // MLP2 raw input (70 -> pad 96)
#define STAGE_HALVES    28672                // largest layer tile: 128 x 224
#define WAVES_PER_BLOCK 4
#define N_RAYS          8192
#define T_STEP          (2.05f / 63.0f)
// per-wave halves: P + Q + IN1 + IN2 = 8704+8704+2048+6144
#define WAVE_HALVES     (64*ACT_PITCH*2 + 64*IN1_PITCH + 64*IN2_PITCH)   // 25600
#define F32_PER_WAVE    448                  // sigma/alpha/cp/w (64 ea) + rgb(192)
// total dynamic LDS: 2*57344 + 4*(51200 + 1792) = 326656 B  (<= 320KB WGP LDS)
#define SMEM_BYTES      (2*STAGE_HALVES*2 + WAVES_PER_BLOCK*(WAVE_HALVES*2 + F32_PER_WAVE*4))

#if defined(__HIP_DEVICE_COMPILE__) && __has_builtin(__builtin_amdgcn_tensor_load_to_lds) && __has_builtin(__builtin_amdgcn_s_wait_tensorcnt)
#define USE_TDM 1
#else
#define USE_TDM 0
#endif

struct LDesc { int wOff, bOff, KcMain, KcAux, Ntiles, srcSel, auxSel, dstSel, kind; };
// srcSel: 0=P 1=Q 2=IN1 3=IN2 ; auxSel: 2=IN1 3=IN2 ; dstSel: 0=P 1=Q
// kind: 0=leaky 1=MLP1-out 2=MLP2-out
__device__ __constant__ LDesc LD[12] = {
  {0,      0,    1, 0, 8, 2, 2, 0, 0},   // M1 L0 : IN1        -> P   (K=32)
  {4096,   128,  4, 0, 8, 0, 2, 1, 0},   // M1 L1 : P          -> Q
  {20480,  256,  4, 1, 8, 1, 2, 0, 0},   // M1 L2 : Q (+IN1)   -> P   (K=160)
  {40960,  384,  4, 0, 8, 0, 2, 1, 0},   // M1 L3 : P          -> Q
  {57344,  512,  4, 0, 8, 1, 2, 0, 0},   // M1 L4 : Q          -> P
  {73728,  640,  4, 0, 5, 0, 2, 0, 1},   // M1 out: P          -> sigma + IN2[0..63]
  {83968,  720,  3, 0, 8, 3, 2, 0, 0},   // M2 L0 : IN2        -> P   (K=96)
  {96256,  848,  4, 0, 8, 0, 2, 1, 0},   // M2 L1 : P          -> Q
  {112640, 976,  4, 3, 8, 1, 3, 0, 0},   // M2 L2 : Q (+IN2)   -> P   (K=224)
  {141312, 1104, 4, 0, 8, 0, 2, 1, 0},   // M2 L3 : P          -> Q
  {157696, 1232, 4, 0, 8, 1, 2, 0, 0},   // M2 L4 : Q          -> P
  {174080, 1360, 4, 0, 1, 0, 2, 0, 2},   // M2 out: P          -> rgb (sigmoid)
};

// --- WMMA fragment loaders (CDNA5 ISA §7.12.2) -------------------
__device__ inline v16h load_a_frag(const h16* __restrict__ S, int pitch, int m, int kc, int lane) {
  const int row   = m * 16 + (lane & 15);
  const int kbase = kc * 32 + ((lane & 16) ? 8 : 0);
  const h16* p = S + row * pitch + kbase;
  v8h lo = *(const v8h*)p;           // K = kbase .. +7
  v8h hi = *(const v8h*)(p + 16);    // K = kbase+16 .. +23
  return __builtin_shufflevector(lo, hi, 0,1,2,3,4,5,6,7,8,9,10,11,12,13,14,15);
}
__device__ inline v16h load_b_frag(const h16* __restrict__ W, int n, int kc, int Kpad, int lane) {
  const int col  = n * 16 + (lane & 15);
  const int koff = kc * 32 + ((lane & 16) ? 8 : 0);
  const h16* p = W + col * Kpad + koff;
  v8h lo = *(const v8h*)p;
  v8h hi = *(const v8h*)(p + 16);
  return __builtin_shufflevector(lo, hi, 0,1,2,3,4,5,6,7,8,9,10,11,12,13,14,15);
}

#if USE_TDM
// Tensor Data Mover: DMA `numHalves` contiguous halves from global -> LDS.
// D# per CDNA5 ISA: count=1, type=2, data_size=4B,
// one-row 2D tile of numHalves/2 dwords. Dynamic LDS starts at offset 0.
__device__ inline void tdm_load(const h16* gsrc, unsigned ldsByteOff, int numHalves) {
  const unsigned long long ga = (unsigned long long)(uintptr_t)gsrc;
  const unsigned dwords = (unsigned)numHalves >> 1;      // <= 14336, fits tile_dim0
  u32x4 g0;
  g0[0] = 1u;                                            // count=1 (user descriptor)
  g0[1] = ldsByteOff;                                    // lds_addr
  g0[2] = (unsigned)(ga & 0xFFFFFFFFu);                  // global_addr[31:0]
  g0[3] = (unsigned)((ga >> 32) & 0x1FFFFFFu) | (2u << 30); // global_addr[56:32], type=2
  i32x8 g1;
  g1[0] = 0x00020000;                                    // data_size=2 (4B), no flags
  g1[1] = (int)((dwords & 0xFFFFu) << 16);               // tensor_dim0[15:0]
  g1[2] = (int)((dwords >> 16) & 0xFFFFu) | (1 << 16);   // tensor_dim0[31:16], tensor_dim1=1
  g1[3] = (int)((dwords & 0xFFFFu) << 16);               // tile_dim0 = dwords
  g1[4] = 1;                                             // tile_dim1 = 1
  g1[5] = (int)dwords;                                   // tensor_dim0_stride[31:0]
  g1[6] = 0; g1[7] = 0;
  const i32x4 z4 = {0, 0, 0, 0};
#if __clang_major__ >= 23
  const i32x8 z8 = {0, 0, 0, 0, 0, 0, 0, 0};
  __builtin_amdgcn_tensor_load_to_lds(g0, g1, z4, z4, z8, 0);
#else
  __builtin_amdgcn_tensor_load_to_lds(g0, g1, z4, z4, 0);
#endif
}
#endif

// --- prep: transpose & f16-convert one weight matrix + pad bias -------------
__global__ void nerf_prep(const float* __restrict__ W, const float* __restrict__ B,
                          h16* __restrict__ wdst, float* __restrict__ bdst,
                          int d_in, int ncols, int Kpad, int ncolpad) {
  const int i = blockIdx.x * blockDim.x + threadIdx.x;
  const int total = ncolpad * Kpad;
  if (i < total) {
    const int n = i / Kpad, k = i % Kpad;
    float v = (k < d_in && n < ncols) ? W[k * ncols + n] : 0.0f;
    wdst[n * Kpad + k] = (h16)v;
  }
  if (i < ncolpad) bdst[i] = (i < ncols) ? B[i] : 0.0f;
}

// --- fused main kernel: 1 wave == 1 ray -------------------------------------
__global__ __launch_bounds__(128) void nerf_fused(
    const float* __restrict__ rays, const float* __restrict__ lights,
    const h16* __restrict__ wgt, const float* __restrict__ bias,
    float* __restrict__ out) {
  extern __shared__ __align__(16) char smem_raw[];
  h16*   stage0  = (h16*)smem_raw;                       // LDS byte offset 0
  h16*   stage1  = stage0 + STAGE_HALVES;                // LDS byte offset 57344
  h16*   actBase = stage1 + STAGE_HALVES;
  float* fBase   = (float*)(actBase + WAVES_PER_BLOCK * WAVE_HALVES);

  const int tid  = threadIdx.x;
  const int wv   = tid >> 5;
  const int lane = tid & 31;

  h16* P   = actBase + wv * WAVE_HALVES;
  h16* Q   = P + 64 * ACT_PITCH;
  h16* IN1 = Q + 64 * ACT_PITCH;
  h16* IN2 = IN1 + 64 * IN1_PITCH;

  float* sigma = fBase + wv * F32_PER_WAVE;
  float* alpha = sigma + 64;
  float* cp    = alpha + 64;
  float* wts   = cp + 64;
  float* rgb   = wts + 64;                               // 64 x 3

  const int r = blockIdx.x * WAVES_PER_BLOCK + wv;
  const float ro0 = rays[r*6+0], ro1 = rays[r*6+1], ro2 = rays[r*6+2];
  const float rd0 = rays[r*6+3], rd1 = rays[r*6+4], rd2 = rays[r*6+5];
  const int   bb  = r >> 12;                             // 4096 rays per batch
  const float li0 = lights[bb*3+0], li1 = lights[bb*3+1], li2 = lights[bb*3+2];

#if USE_TDM
  if (wv == 0)  // prefetch layer 0 weights into stage0 while everyone inits
    tdm_load(wgt + LD[0].wOff, 0u, LD[0].Ntiles * 16 * ((LD[0].KcMain + LD[0].KcAux) << 5));
#endif

  // Zero IN1+IN2 (contiguous: 2048+6144 halves = 16384 B), then fill inputs.
  {
    u32x4 z = {0u, 0u, 0u, 0u};
    for (int i = lane; i < 1024; i += 32) *(u32x4*)(IN1 + i * 8) = z;
  }
  for (int s = lane; s < 64; s += 32) {
    const float t = s * T_STEP;
    h16* r1 = IN1 + s * IN1_PITCH;                       // pts = r_o + t*r_d
    r1[0] = (h16)(ro0 + t * rd0);
    r1[1] = (h16)(ro1 + t * rd1);
    r1[2] = (h16)(ro2 + t * rd2);
    h16* r2 = IN2 + s * IN2_PITCH;                       // cols 64..69: r_d, light
    r2[64] = (h16)rd0; r2[65] = (h16)rd1; r2[66] = (h16)rd2;
    r2[67] = (h16)li0; r2[68] = (h16)li1; r2[69] = (h16)li2;
  }

  for (int L = 0; L < 12; L++) {
    const LDesc d = LD[L];
    const int Kpad = (d.KcMain + d.KcAux) << 5;
    h16* wbuf = (L & 1) ? stage1 : stage0;

    __syncthreads();   // everyone done reading the stage buffer being replaced
#if USE_TDM
    if (wv == 0) {
      if (L < 11) {    // kick off next layer's DMA, then wait only for current
        const LDesc nd = LD[L + 1];
        tdm_load(wgt + nd.wOff, ((L + 1) & 1) ? (unsigned)(STAGE_HALVES * 2) : 0u,
                 nd.Ntiles * 16 * ((nd.KcMain + nd.KcAux) << 5));
        __builtin_amdgcn_s_wait_tensorcnt(1);
      } else {
        __builtin_amdgcn_s_wait_tensorcnt(0);
      }
    }
#else
    {                  // manual staging fallback
      const u32x4* src = (const u32x4*)(wgt + d.wOff);
      u32x4* dst = (u32x4*)wbuf;
      const int n16 = (d.Ntiles * 16 * Kpad) / 8;
      for (int i = tid; i < n16; i += 128) dst[i] = src[i];
    }
#endif
    __syncthreads();   // stage buffer for layer L ready for all waves

    // Ternary selection (NOT arrays): keeps pointers provably in LDS space.
    const h16* S  = (d.srcSel == 0) ? P : (d.srcSel == 1) ? Q
                   : (d.srcSel == 2) ? IN1 : IN2;
    const int  sP = (d.srcSel < 2) ? ACT_PITCH
                   : (d.srcSel == 2) ? IN1_PITCH : IN2_PITCH;
    const h16* AX = (d.auxSel == 2) ? IN1 : IN2;
    const int  aP = (d.auxSel == 2) ? IN1_PITCH : IN2_PITCH;
    h16*       D  = (d.dstSel == 0) ? P : Q;
    const float* bptr = bias + d.bOff;

    for (int n = 0; n < d.Ntiles; n++) {
      const int col = n * 16 + (lane & 15);
      const float bv = bptr[col];
      v8f acc[4];
      #pragma unroll
      for (int m = 0; m < 4; m++) {
        #pragma unroll
        for (int v = 0; v < 8; v++) acc[m][v] = bv;
      }
      for (int kc = 0; kc < d.KcMain; kc++) {            // main K-pass
        const v16h bf = load_b_frag(wbuf, n, kc, Kpad, lane);
        #pragma unroll
        for (int m = 0; m < 4; m++) {
          const v16h af = load_a_frag(S, sP, m, kc, lane);
          acc[m] = __builtin_amdgcn_wmma_f32_16x16x32_f16(
              false, af, false, bf, (short)0, acc[m], false, false);
        }
      }
      for (int kc = 0; kc < d.KcAux; kc++) {             // skip-concat K-pass
        const v16h bf = load_b_frag(wbuf, n, d.KcMain + kc, Kpad, lane);
        #pragma unroll
        for (int m = 0; m < 4; m++) {
          const v16h af = load_a_frag(AX, aP, m, kc, lane);
          acc[m] = __builtin_amdgcn_wmma_f32_16x16x32_f16(
              false, af, false, bf, (short)0, acc[m], false, false);
        }
      }
      if (d.kind == 0) {                                 // leaky_relu(0.01)
        #pragma unroll
        for (int m = 0; m < 4; m++) {
          #pragma unroll
          for (int v = 0; v < 8; v++) {
            float x = acc[m][v];
            x = x > 0.0f ? x : 0.01f * x;
            const int row = m * 16 + v + ((lane & 16) ? 8 : 0);
            D[row * ACT_PITCH + col] = (h16)x;
          }
        }
      } else if (d.kind == 1) {                          // sigma + latent -> IN2
        #pragma unroll
        for (int m = 0; m < 4; m++) {
          #pragma unroll
          for (int v = 0; v < 8; v++) {
            const float x = acc[m][v];
            const int row = m * 16 + v + ((lane & 16) ? 8 : 0);
            if (col == 0)          sigma[row] = x;       // raw; relu in render
            else if (col < 65)     IN2[row * IN2_PITCH + (col - 1)] = (h16)x;
          }
        }
      } else {                                           // sigmoid rgb
        #pragma unroll
        for (int m = 0; m < 4; m++) {
          #pragma unroll
          for (int v = 0; v < 8; v++) {
            const int row = m * 16 + v + ((lane & 16) ? 8 : 0);
            if (col < 3) {
              const float x = acc[m][v];
              rgb[row * 3 + col] = 1.0f / (1.0f + __expf(-x));
            }
          }
        }
      }
    }
  }

  // --- volume rendering (per-wave, LDS) ------------------------------------
  for (int s = lane; s < 64; s += 32) {
    const float t  = s * T_STEP;
    const float sg = fmaxf(sigma[s], 0.0f);
    alpha[s] = 1.0f - __expf(-sg * t);
  }
  if (lane == 0) {
    float run = 1.0f;
    for (int s = 0; s < 64; s++) {
      run *= fmaxf(1.0f - alpha[s], 1e-10f);
      cp[s] = run;
    }
    const float last = cp[63];
    for (int s = 0; s < 64; s++) {
      const float c = (s == 0) ? last : ((s == 63) ? 1.0f : cp[s - 1]);
      wts[s] = alpha[s] * c;     // matches reference roll(1) + at[-1].set(1)
    }
  }
  if (lane < 3) {
    float accum = 0.0f;
    for (int s = 0; s < 64; s++) accum += wts[s] * rgb[s * 3 + lane];
    out[r * 3 + lane] = accum;
  }
}

// ---------------------------------------------------------------------------
extern "C" void kernel_launch(void* const* d_in, const int* in_sizes, int n_in,
                              void* d_out, int out_size, void* d_ws, size_t ws_size,
                              hipStream_t stream) {
  (void)in_sizes; (void)n_in; (void)out_size; (void)ws_size;
  // d_in order: rays, lights, first_Ws[0..5], first_bs[0..5], second_Ws[0..5], second_bs[0..5]
  const float* rays   = (const float*)d_in[0];
  const float* lights = (const float*)d_in[1];
  h16*   wgt     = (h16*)d_ws;                       // 176128 halves = 352256 B
  float* biasBuf = (float*)((char*)d_ws + 352256);   // 1376 floats

  struct PT { int din, ncols, Kpad, ncolpad, wOff, bOff, iW, iB; };
  const PT T[12] = {
    {3,   128, 32,  128, 0,      0,    2,  8},
    {128, 128, 128, 128, 4096,   128,  3,  9},
    {131, 128, 160, 128, 20480,  256,  4,  10},
    {128, 128, 128, 128, 40960,  384,  5,  11},
    {128, 128, 128, 128, 57344,  512,  6,  12},
    {128, 65,  128, 80,  73728,  640,  7,  13},
    {70,  128, 96,  128, 83968,  720,  14, 20},
    {128, 128, 128, 128, 96256,  848,  15, 21},
    {198, 128, 224, 128, 112640, 976,  16, 22},
    {128, 128, 128, 128, 141312, 1104, 17, 23},
    {128, 128, 128, 128, 157696, 1232, 18, 24},
    {128, 3,   128, 16,  174080, 1360, 19, 25},
  };
  for (int l = 0; l < 12; l++) {
    const int total  = T[l].ncolpad * T[l].Kpad;
    const int blocks = (total + 255) / 256;
    hipLaunchKernelGGL(nerf_prep, dim3(blocks), dim3(256), 0, stream,
                       (const float*)d_in[T[l].iW], (const float*)d_in[T[l].iB],
                       wgt + T[l].wOff, biasBuf + T[l].bOff,
                       T[l].din, T[l].ncols, T[l].Kpad, T[l].ncolpad);
  }

  hipLaunchKernelGGL(nerf_fused, dim3(N_RAYS / WAVES_PER_BLOCK), dim3(128),
                     SMEM_BYTES, stream,
                     rays, lights, (const h16*)wgt, (const float*)biasBuf, (float*)d_out);
}
